// AMSoftmax_normfree_74019466379413
// MI455X (gfx1250) — matrix operand order repaired and checked by reading, not compiled, not run on hardware
//
#include <hip/hip_runtime.h>

typedef __attribute__((ext_vector_type(16))) _Float16 v16h;
typedef __attribute__((ext_vector_type(8)))  _Float16 v8h;
typedef __attribute__((ext_vector_type(4)))  _Float16 v4h;
typedef __attribute__((ext_vector_type(8)))  float    v8f;

#define BB 4096   // batch rows
#define DD 512    // inner dim (K)
#define CC 8000   // classes (N)

// ---------------------------------------------------------------------------
// Kernel 1: per-row L2 normalize (f32 -> f16), one wave32 per row of 512.
// Optionally writes the raw row norm (needed for logits scaling).
// ---------------------------------------------------------------------------
__global__ __launch_bounds__(256) void rownorm_f16(
    const float* __restrict__ src, _Float16* __restrict__ dst,
    float* __restrict__ normOut, int nrows)
{
  const int row  = blockIdx.x * (blockDim.x >> 5) + (threadIdx.x >> 5);
  const int lane = threadIdx.x & 31;
  if (row >= nrows) return;

  const float4* rp = (const float4*)(src + (size_t)row * DD);
  float4 v[4];
  float ss = 0.f;
#pragma unroll
  for (int j = 0; j < 4; ++j) {
    v[j] = rp[lane + j * 32];            // cols lane*4 + j*128 .. +3 (coalesced)
    ss += v[j].x * v[j].x + v[j].y * v[j].y + v[j].z * v[j].z + v[j].w * v[j].w;
  }
#pragma unroll
  for (int off = 16; off > 0; off >>= 1)
    ss += __shfl_xor(ss, off, 32);

  const float nrm = sqrtf(ss);
  const float r   = 1.f / fmaxf(nrm, 1e-12f);
  if (normOut && lane == 0) normOut[row] = nrm;

  _Float16* dp = dst + (size_t)row * DD;
#pragma unroll
  for (int j = 0; j < 4; ++j) {
    v4h h;
    h[0] = (_Float16)(v[j].x * r);
    h[1] = (_Float16)(v[j].y * r);
    h[2] = (_Float16)(v[j].z * r);
    h[3] = (_Float16)(v[j].w * r);
    *(v4h*)(dp + (size_t)(lane + j * 32) * 4) = h;
  }
}

// ---------------------------------------------------------------------------
// Kernel 2: WMMA GEMM  cosine = xhat @ what^T  with fused AM-Softmax epilogue.
// Block = 256 threads = 8 waves. Tile 128(M) x 128(N), BK = 32.
// Wave grid 4(M) x 2(N); each wave computes 2x4 = 8 wmma tiles (32x64 region).
// ---------------------------------------------------------------------------
static __device__ inline v16h mk16(v8h lo, v8h hi)
{
  v16h r;
#pragma unroll
  for (int i = 0; i < 8; ++i) { r[i] = lo[i]; r[i + 8] = hi[i]; }
  return r;
}

__global__ __launch_bounds__(256) void amsm_gemm(
    const _Float16* __restrict__ xh, const _Float16* __restrict__ wh,
    const float* __restrict__ normv, const int* __restrict__ label,
    const float* __restrict__ mptr,
    float* __restrict__ logits, float* __restrict__ cosine)
{
  // 40-half row stride (80B): bank-conflict padding, keeps 16B alignment.
  __shared__ __align__(16) _Float16 As[128][40];
  __shared__ __align__(16) _Float16 Bs[128][40];

  const int tid  = threadIdx.x;
  const int lane = tid & 31;
  const int wave = tid >> 5;      // 0..7
  const int wm   = wave >> 1;     // 0..3  -> 32-row slab
  const int wn   = wave & 1;      // 0..1  -> 64-col slab
  const int lr   = lane & 15;
  const int lh   = lane >> 4;     // half-wave selector

  const int n_base = blockIdx.x * 128;
  const int m_base = blockIdx.y * 128;

  const float mval = mptr[0];

  v8f acc[2][4];
  {
    v8f z = {};
#pragma unroll
    for (int i = 0; i < 2; ++i)
#pragma unroll
      for (int j = 0; j < 4; ++j) acc[i][j] = z;
  }

  // cooperative tile loaders: 2 threads per row, 16 halves (32B) each
  const int ldr = tid >> 1;            // 0..127
  const int ldh = (tid & 1) * 16;      // 0 or 16 (halves)
  const _Float16* gA = xh + (size_t)(m_base + ldr) * DD + ldh;
  const int grB = n_base + ldr;
  const bool bvalid = grB < CC;
  const _Float16* gB = wh + (size_t)grB * DD + ldh;

  for (int k0 = 0; k0 < DD; k0 += 32) {
    v8h a0 = *(const v8h*)(gA + k0);
    v8h a1 = *(const v8h*)(gA + k0 + 8);
    v8h b0 = {}, b1 = {};
    if (bvalid) {
      b0 = *(const v8h*)(gB + k0);
      b1 = *(const v8h*)(gB + k0 + 8);
    }

    __syncthreads();                       // previous compute done
    *(v8h*)&As[ldr][ldh]     = a0;
    *(v8h*)&As[ldr][ldh + 8] = a1;
    *(v8h*)&Bs[ldr][ldh]     = b0;
    *(v8h*)&Bs[ldr][ldh + 8] = b1;
    __syncthreads();                       // tile visible

    // A fragments (ISA 16-bit A 16x32: low lanes K 0-7|16-23, high lanes K 8-15|24-31)
    v16h af[2];
#pragma unroll
    for (int mt = 0; mt < 2; ++mt) {
      const _Float16* ap = &As[wm * 32 + mt * 16 + lr][lh * 8];
      af[mt] = mk16(*(const v8h*)ap, *(const v8h*)(ap + 16));
    }
    // B fragments (low lanes K 0-7|16-23, high lanes K 8-15|24-31, N = lane%16)
    v16h bf[4];
#pragma unroll
    for (int nt = 0; nt < 4; ++nt) {
      const _Float16* bp = &Bs[wn * 64 + nt * 16 + lr][lh * 8];
      bf[nt] = mk16(*(const v8h*)bp, *(const v8h*)(bp + 16));
    }

#pragma unroll
    for (int mt = 0; mt < 2; ++mt)
#pragma unroll
      for (int nt = 0; nt < 4; ++nt)
        acc[mt][nt] = __builtin_amdgcn_wmma_f32_16x16x32_f16(
            false, af[mt], false, bf[nt], (short)0, acc[mt][nt], false, false);
  }

  // Epilogue. C/D layout: VGPR e -> M = e (lanes 0-15) / 8+e (lanes 16-31); N = lane%16.
#pragma unroll
  for (int mt = 0; mt < 2; ++mt) {
#pragma unroll
    for (int e = 0; e < 8; ++e) {
      const int row = m_base + wm * 32 + mt * 16 + lh * 8 + e;  // always < BB
      const float nr = normv[row];
      const int  lb = label[row];
#pragma unroll
      for (int nt = 0; nt < 4; ++nt) {
        const int col = n_base + wn * 64 + nt * 16 + lr;
        if (col < CC) {
          const float v = acc[mt][nt][e];
          const size_t idx = (size_t)row * CC + col;
          cosine[idx] = v;
          const float mg = (lb == col) ? mval : 0.f;
          logits[idx] = nr * (v - mg);
        }
      }
    }
  }
}

// ---------------------------------------------------------------------------
// Launcher
// ---------------------------------------------------------------------------
extern "C" void kernel_launch(void* const* d_in, const int* in_sizes, int n_in,
                              void* d_out, int out_size, void* d_ws, size_t ws_size,
                              hipStream_t stream)
{
  (void)in_sizes; (void)n_in; (void)out_size; (void)ws_size;

  const float* input  = (const float*)d_in[0];   // [4096,512] f32
  const int*   label  = (const int*)  d_in[1];   // [4096] i32
  const float* weight = (const float*)d_in[2];   // [8000,512] f32
  /* d_in[3] = s : unused in this branch (faithful to reference) */
  const float* mptr   = (const float*)d_in[4];   // scalar m

  float* logits = (float*)d_out;                         // [4096,8000]
  float* cosine = logits + (size_t)BB * CC;              // [4096,8000]

  // workspace: norms (16 KB) | xhat f16 (4 MB) | what f16 (8 MB)  ~ 12.2 MB
  float*    normv = (float*)d_ws;
  _Float16* xh = (_Float16*)((char*)d_ws + BB * sizeof(float));
  _Float16* wh = xh + (size_t)BB * DD;

  rownorm_f16<<<dim3(BB / 8), 256, 0, stream>>>(input,  xh, normv,   BB);
  rownorm_f16<<<dim3(CC / 8), 256, 0, stream>>>(weight, wh, nullptr, CC);
  amsm_gemm<<<dim3((CC + 127) / 128, BB / 128), 256, 0, stream>>>(
      xh, wh, normv, label, mptr, logits, cosine);
}
